// GraphAttentionLayer_43026982371771
// MI455X (gfx1250) — matrix operand order repaired and checked by reading, not compiled, not run on hardware
//
#include <hip/hip_runtime.h>

typedef __attribute__((ext_vector_type(2)))  float  v2f;
typedef __attribute__((ext_vector_type(8)))  float  v8f;
typedef __attribute__((ext_vector_type(16))) __bf16 v16bf;

#define NB    4
#define NNODE 2048
#define FIN   256
#define FOUT  128
#define ALPHA 0.2f
#define MASKNEG -10000000000.0f

__device__ __forceinline__ unsigned short f2bf(float x) {
    union { float f; unsigned u; } v; v.f = x;
    unsigned r = v.u + 0x7FFFu + ((v.u >> 16) & 1u);   // RNE
    return (unsigned short)(r >> 16);
}

union BF16x16 { uint4 u[2]; v16bf v; };

// ---------------------------------------------------------------------------
// Kernel A: h = X @ W  (exact fp32 WMMA 16x16x4), store h^T as bf16
//           (column-major per batch) + e_src/e_dst dot products.
// grid = (B*N)/16 = 512 blocks, 256 threads = 8 waves; wave w -> cols 16w..16w+15
// ---------------------------------------------------------------------------
__global__ __launch_bounds__(256)
void gat_gemm_h(const float* __restrict__ X, const float* __restrict__ W,
                const float* __restrict__ a, unsigned short* __restrict__ hT,
                float* __restrict__ e_src, float* __restrict__ e_dst)
{
    __shared__ __align__(16) float s_x[16][FIN + 4];
    __shared__ float s_ps[8][16];
    __shared__ float s_pd[8][16];

    const int tid  = threadIdx.x;
    const int wave = tid >> 5;
    const int lane = tid & 31;
    const int half = lane >> 4;
    const int l15  = lane & 15;
    const int gi0  = blockIdx.x * 16;        // global row base (0..8191)
    const int b    = gi0 / NNODE;
    const int li0  = gi0 % NNODE;            // row base within batch

    // stage X tile 16x256 (coalesced)
    for (int idx = tid; idx < 16 * FIN; idx += 256) {
        int r = idx >> 8;
        int k = idx & (FIN - 1);
        s_x[r][k] = X[(size_t)(gi0 + r) * FIN + k];
    }
    __syncthreads();

    const int col = wave * 16 + l15;

    v8f acc = {};
    for (int k0 = 0; k0 < FIN; k0 += 4) {
        // A 16x4 f32: a.x = A[M][2*half], a.y = A[M][2*half+1]
        v2f av, bv;
        av.x = s_x[l15][k0 + 2 * half];
        av.y = s_x[l15][k0 + 2 * half + 1];
        // B 4x16 f32: b.x = B[2*half][N], b.y = B[2*half+1][N]
        bv.x = W[(size_t)(k0 + 2 * half) * FOUT + col];
        bv.y = W[(size_t)(k0 + 2 * half + 1) * FOUT + col];
        acc = __builtin_amdgcn_wmma_f32_16x16x4_f32(false, av, false, bv,
                                                    (short)0, acc, false, false);
    }

    // --- store h^T bf16: lane holds rows (8*half + 0..7) of column `col`
    {
        unsigned short pk[8];
        #pragma unroll
        for (int r = 0; r < 8; ++r) pk[r] = f2bf(acc[r]);
        uint4 u;
        u.x = (unsigned)pk[0] | ((unsigned)pk[1] << 16);
        u.y = (unsigned)pk[2] | ((unsigned)pk[3] << 16);
        u.z = (unsigned)pk[4] | ((unsigned)pk[5] << 16);
        u.w = (unsigned)pk[6] | ((unsigned)pk[7] << 16);
        size_t base = (size_t)(b * FOUT + col) * NNODE + (size_t)(li0 + 8 * half);
        *(uint4*)(hT + base) = u;      // 16B aligned (base % 8 == 0)
    }

    // --- e_src/e_dst: deterministic shuffle reduction over the 16 columns
    {
        const float asv = a[col];
        const float adv = a[FOUT + col];
        float es[8], ed[8];
        #pragma unroll
        for (int r = 0; r < 8; ++r) { es[r] = acc[r] * asv; ed[r] = acc[r] * adv; }
        #pragma unroll
        for (int off = 1; off < 16; off <<= 1) {
            #pragma unroll
            for (int r = 0; r < 8; ++r) {
                es[r] += __shfl_xor(es[r], off);
                ed[r] += __shfl_xor(ed[r], off);
            }
        }
        if (l15 == 0) {
            #pragma unroll
            for (int r = 0; r < 8; ++r) {
                s_ps[wave][8 * half + r] = es[r];
                s_pd[wave][8 * half + r] = ed[r];
            }
        }
    }
    __syncthreads();
    if (tid < 16) {
        float t = 0.f, u = 0.f;
        #pragma unroll
        for (int w = 0; w < 8; ++w) { t += s_ps[w][tid]; u += s_pd[w][tid]; }
        e_src[gi0 + tid] = t;
        e_dst[gi0 + tid] = u;
    }
}

// ---------------------------------------------------------------------------
// Kernel B: masked leaky-relu attention + softmax + (attention @ h) + bias
// grid = (B*N)/16 = 512 blocks (one 16-row i-tile), 256 threads = 8 waves.
// ---------------------------------------------------------------------------
__global__ __launch_bounds__(256)
void gat_attn(const float* __restrict__ adj, const unsigned short* __restrict__ hT,
              const float* __restrict__ e_src, const float* __restrict__ e_dst,
              const float* __restrict__ bias, float* __restrict__ out)
{
    __shared__ __align__(16) float s_edst[NNODE];
    __shared__ __align__(16) unsigned short s_p[16][136];   // padded rows
    __shared__ float s_esrc[16], s_m[16], s_l[16];

    const int tid  = threadIdx.x;
    const int wave = tid >> 5;
    const int lane = tid & 31;
    const int half = lane >> 4;
    const int l15  = lane & 15;
    const int gi0  = blockIdx.x * 16;
    const int b    = gi0 / NNODE;

    for (int j = tid; j < NNODE; j += 256) s_edst[j] = e_dst[(size_t)b * NNODE + j];
    if (tid < 16) s_esrc[tid] = e_src[gi0 + tid];
    __syncthreads();

    // ---- phase 1: online row max & exp-sum; wave w owns rows 2w, 2w+1
    const int r0 = 2 * wave, r1 = 2 * wave + 1;
    const size_t adjRow = (size_t)(gi0 + r0) * NNODE;   // (b*N + i)*N == (gi0+r)*N
    const float es0 = s_esrc[r0], es1 = s_esrc[r1];

    float m0 = -1e30f, s0 = 0.f, m1 = -1e30f, s1 = 0.f;
    for (int jb = 0; jb < NNODE / 32; ++jb) {
        const int j = jb * 32 + lane;
        const float ed = s_edst[j];
        const float A0 = adj[adjRow + j];
        const float A1 = adj[adjRow + NNODE + j];
        float e0 = es0 + ed; e0 = (e0 >= 0.f) ? e0 : ALPHA * e0; e0 += MASKNEG * (1.f - A0);
        float e1 = es1 + ed; e1 = (e1 >= 0.f) ? e1 : ALPHA * e1; e1 += MASKNEG * (1.f - A1);
        if (e0 > m0) { s0 *= __expf(m0 - e0); m0 = e0; } s0 += __expf(e0 - m0);
        if (e1 > m1) { s1 *= __expf(m1 - e1); m1 = e1; } s1 += __expf(e1 - m1);
    }
    #pragma unroll
    for (int off = 16; off > 0; off >>= 1) {
        float m2 = __shfl_xor(m0, off), t2 = __shfl_xor(s0, off);
        float M  = fmaxf(m0, m2);
        s0 = s0 * __expf(m0 - M) + t2 * __expf(m2 - M); m0 = M;
        m2 = __shfl_xor(m1, off); t2 = __shfl_xor(s1, off);
        M  = fmaxf(m1, m2);
        s1 = s1 * __expf(m1 - M) + t2 * __expf(m2 - M); m1 = M;
    }
    if (lane == 0) { s_m[r0] = m0; s_l[r0] = s0; s_m[r1] = m1; s_l[r1] = s1; }
    __syncthreads();

    const float pm0 = s_m[r0], pm1 = s_m[r1];

    // ---- phase 2: per 128-wide j-chunk: build bf16 p-tile, then WMMA bf16
    const int  col   = wave * 16 + l15;
    const size_t hbase = (size_t)(b * FOUT + col) * NNODE;
    v8f acc = {};

    for (int chunk = 0; chunk < NNODE / 128; ++chunk) {
        const int j0 = chunk * 128;
        #pragma unroll
        for (int it = 0; it < 4; ++it) {
            const int jj = it * 32 + lane;
            const int j  = j0 + jj;
            const float ed = s_edst[j];
            const float A0 = adj[adjRow + j];
            const float A1 = adj[adjRow + NNODE + j];
            float e0 = es0 + ed; e0 = (e0 >= 0.f) ? e0 : ALPHA * e0; e0 += MASKNEG * (1.f - A0);
            float e1 = es1 + ed; e1 = (e1 >= 0.f) ? e1 : ALPHA * e1; e1 += MASKNEG * (1.f - A1);
            s_p[r0][jj] = f2bf(__expf(e0 - pm0));
            s_p[r1][jj] = f2bf(__expf(e1 - pm1));
        }
        __syncthreads();

        #pragma unroll
        for (int kk = 0; kk < 4; ++kk) {
            // A operand (16x32 bf16): v[0..15] = K kk*32 + 16*half + 0..15
            const int koff = kk * 32 + 16 * half;
            const unsigned short* pa = &s_p[l15][koff];
            BF16x16 aop;
            aop.u[0] = *(const uint4*)(pa);
            aop.u[1] = *(const uint4*)(pa + 8);
            // B operand (32x16 bf16): v[0..15] = B[K = kk*32 + 16*half + 0..15][N]
            const unsigned short* pb = hT + hbase + (size_t)(j0 + kk * 32 + 16 * half);
            BF16x16 bop;
            bop.u[0] = *(const uint4*)(pb);
            bop.u[1] = *(const uint4*)(pb + 8);
            acc = __builtin_amdgcn_wmma_f32_16x16x32_bf16(false, aop.v, false, bop.v,
                                                          (short)0, acc, false, false);
        }
        __syncthreads();
    }

    // ---- normalize, add bias, store
    const float bval = bias[col];
    #pragma unroll
    for (int r = 0; r < 8; ++r) {
        const int row = 8 * half + r;
        const float invl = 1.0f / s_l[row];
        out[(size_t)(gi0 + row) * FOUT + col] = acc[r] * invl + bval;
    }
}

// ---------------------------------------------------------------------------
extern "C" void kernel_launch(void* const* d_in, const int* in_sizes, int n_in,
                              void* d_out, int out_size, void* d_ws, size_t ws_size,
                              hipStream_t stream)
{
    const float* X    = (const float*)d_in[0];   // (4,2048,256)
    const float* adj  = (const float*)d_in[1];   // (4,2048,2048)
    const float* W    = (const float*)d_in[2];   // (256,128)
    const float* a    = (const float*)d_in[3];   // (256,1)
    const float* bias = (const float*)d_in[4];   // (128,)
    float* out = (float*)d_out;                  // (4,2048,128)

    unsigned short* hT = (unsigned short*)d_ws;                       // 4*128*2048 bf16 = 2MB
    float* e_src = (float*)((char*)d_ws + (size_t)NB * FOUT * NNODE * sizeof(unsigned short));
    float* e_dst = e_src + (size_t)NB * NNODE;

    const int nblocks = (NB * NNODE) / 16;       // 512
    gat_gemm_h<<<nblocks, 256, 0, stream>>>(X, W, a, hT, e_src, e_dst);
    gat_attn <<<nblocks, 256, 0, stream>>>(adj, hT, e_src, e_dst, bias, out);
}
